// SVDHead_4140348473561
// MI455X (gfx1250) — matrix operand order repaired and checked by reading, not compiled, not run on hardware
//
#include <hip/hip_runtime.h>

#define B_  32
#define DK_ 64
#define N_  2048
#define M_  2048

typedef __attribute__((ext_vector_type(16))) _Float16 v16h;
typedef __attribute__((ext_vector_type(8)))  _Float16 v8h;
typedef __attribute__((ext_vector_type(8)))  float    v8f;

__device__ __forceinline__ v8f wmma16(v16h a, v16h b, v8f c) {
  // D = A(16x32 f16) * B(32x16 f16) + C(16x16 f32)
  return __builtin_amdgcn_wmma_f32_16x16x32_f16(false, a, false, b, (short)0, c, false, false);
}

// ---------------- prep kernels ----------------
__global__ __launch_bounds__(256) void zero_acc_kernel(float* acc) {
  int i = blockIdx.x * 256 + threadIdx.x;
  if (i < B_ * 16) acc[i] = 0.0f;
}

// action_embedding [B][DK][N] f32  ->  Qh [B][N][DK] f16, pre-scaled by 1/sqrt(64)
__global__ __launch_bounds__(256) void conv_q_kernel(const float* __restrict__ in,
                                                     _Float16* __restrict__ out) {
  int idx = blockIdx.x * 256 + threadIdx.x;
  if (idx >= B_ * N_ * DK_) return;
  int d = idx & (DK_ - 1);
  int n = (idx >> 6) & (N_ - 1);
  int b = idx >> 17;
  out[idx] = (_Float16)(in[((size_t)b * DK_ + d) * N_ + n] * 0.125f);
}

// anchor_embedding [B][DK][M] f32 -> Kh [B][M][DK] f16
__global__ __launch_bounds__(256) void conv_k_kernel(const float* __restrict__ in,
                                                     _Float16* __restrict__ out) {
  int idx = blockIdx.x * 256 + threadIdx.x;
  if (idx >= B_ * M_ * DK_) return;
  int d = idx & (DK_ - 1);
  int m = (idx >> 6) & (M_ - 1);
  int b = idx >> 17;
  out[idx] = (_Float16)in[((size_t)b * DK_ + d) * M_ + m];
}

// anchor_points [B][3][M] f32 -> Vt [B][M][4] f32 (padded rows)
__global__ __launch_bounds__(256) void conv_v_kernel(const float* __restrict__ in,
                                                     float* __restrict__ out) {
  int idx = blockIdx.x * 256 + threadIdx.x;
  if (idx >= B_ * M_) return;
  int m = idx & (M_ - 1);
  int b = idx >> 11;
  const float* p = in + (size_t)b * 3 * M_ + m;
  float4 r; r.x = p[0]; r.y = p[M_]; r.z = p[2 * M_]; r.w = 0.0f;
  *(float4*)(out + (size_t)idx * 4) = r;
}

// ---------------- fused attention + covariance accumulation ----------------
// One wave handles one 16-column n-tile of one batch, loops over all m in chunks
// of 16 rows. D tile = scores^T : row M = m (VGPR r + 8*half), col N = n (lane&15).
// Online softmax fully in-lane; final half-merge via one shfl_xor(16).
__global__ __launch_bounds__(256) void attn_acc_kernel(
    const _Float16* __restrict__ Qh, const _Float16* __restrict__ Kh,
    const float* __restrict__ Vt, const float* __restrict__ act_pts,
    float* __restrict__ acc_g)
{
  const int lane  = threadIdx.x & 31;
  const int wave  = threadIdx.x >> 5;
  const int w     = blockIdx.x * 8 + wave;      // 4096 waves total
  const int b     = w >> 7;                     // 128 n-tiles per batch
  const int ncol0 = (w & 127) << 4;
  const int half  = lane >> 4;
  const int lw    = lane & 15;

  // B-fragment (action / Q, fixed per tile): lane col n = ncol0+lw,
  // frag0 holds d = 16*half + e, frag1 holds d = 32 + 16*half + e
  const _Float16* qb = Qh + ((size_t)b * N_ + ncol0 + lw) * DK_;
  v16h q0 = *(const v16h*)(qb + 16 * half);
  v16h q1 = *(const v16h*)(qb + 32 + 16 * half);

  const _Float16* kb0 = Kh + ((size_t)b * M_ + lw) * DK_;
  const float*    vb  = Vt + ((size_t)b * M_ + 8 * half) * 4;

  float mx = -3.0e38f, sm = 0.0f, a0 = 0.0f, a1 = 0.0f, a2 = 0.0f;

  #pragma unroll 1
  for (int mc = 0; mc < M_; mc += 16) {
    // A-fragment (anchor / K): lane row m = mc+lw; contiguous k order, same as q0/q1:
    // e=0..15 -> d = 16*half + e; frag1 offsets +32.
    const _Float16* kp = kb0 + (size_t)mc * DK_;
    v8h kA0 = *(const v8h*)(kp + 16 * half);
    v8h kA1 = *(const v8h*)(kp + 8 + 16 * half);
    v8h kB0 = *(const v8h*)(kp + 32 + 16 * half);
    v8h kB1 = *(const v8h*)(kp + 40 + 16 * half);
    v16h ka = __builtin_shufflevector(kA0, kA1, 0,1,2,3,4,5,6,7,8,9,10,11,12,13,14,15);
    v16h kb = __builtin_shufflevector(kB0, kB1, 0,1,2,3,4,5,6,7,8,9,10,11,12,13,14,15);

    v8f c = {};
    c = wmma16(ka, q0, c);
    c = wmma16(kb, q1, c);   // c[r] = scores^T[mc + r + 8*half][ncol0+lw] (pre-scaled)

    float m8 = fmaxf(fmaxf(fmaxf(c[0], c[1]), fmaxf(c[2], c[3])),
                     fmaxf(fmaxf(c[4], c[5]), fmaxf(c[6], c[7])));
    float nm = fmaxf(mx, m8);
    float f  = __expf(mx - nm);
    float ps = 0.0f, pv0 = 0.0f, pv1 = 0.0f, pv2 = 0.0f;
    const float* vp = vb + (size_t)mc * 4;
    #pragma unroll
    for (int r = 0; r < 8; ++r) {
      float p = __expf(c[r] - nm);
      ps += p;
      float4 vv = *(const float4*)(vp + r * 4);  // V row m = mc + 8*half + r
      pv0 = fmaf(p, vv.x, pv0);
      pv1 = fmaf(p, vv.y, pv1);
      pv2 = fmaf(p, vv.z, pv2);
    }
    sm = fmaf(sm, f, ps);
    a0 = fmaf(a0, f, pv0);
    a1 = fmaf(a1, f, pv1);
    a2 = fmaf(a2, f, pv2);
    mx = nm;
  }

  // merge the two m-halves (lanes l and l+16 hold the same column n)
  {
    float omx = __shfl_xor(mx, 16, 32);
    float osm = __shfl_xor(sm, 16, 32);
    float o0  = __shfl_xor(a0, 16, 32);
    float o1  = __shfl_xor(a1, 16, 32);
    float o2  = __shfl_xor(a2, 16, 32);
    float nm  = fmaxf(mx, omx);
    float e1  = __expf(mx - nm), e2 = __expf(omx - nm);
    sm = sm * e1 + osm * e2;
    a0 = a0 * e1 + o0 * e2;
    a1 = a1 * e1 + o1 * e2;
    a2 = a2 * e1 + o2 * e2;
  }
  float inv = 1.0f / sm;
  float c0 = a0 * inv, c1 = a1 * inv, c2 = a2 * inv;   // action_corr[:, n]

  // per-batch accumulators: 9 of sum a_i*c_j, 3 of sum a_i, 3 of sum c_j
  float msk = (half == 0) ? 1.0f : 0.0f;               // count each column once
  const float* ap = act_pts + (size_t)b * 3 * N_ + ncol0 + lw;
  float x0 = ap[0] * msk, x1 = ap[N_] * msk, x2 = ap[2 * N_] * msk;
  float vals[15] = { x0*c0, x0*c1, x0*c2,
                     x1*c0, x1*c1, x1*c2,
                     x2*c0, x2*c1, x2*c2,
                     x0, x1, x2,
                     c0*msk, c1*msk, c2*msk };
  #pragma unroll
  for (int i = 0; i < 15; ++i) {
    float v = vals[i];
    for (int off = 1; off < 32; off <<= 1) v += __shfl_xor(v, off, 32);
    vals[i] = v;
  }
  if (lane == 0) {
    float* dst = acc_g + b * 16;
    #pragma unroll
    for (int i = 0; i < 15; ++i) atomicAdd(dst + i, vals[i]);
  }
}

// ---------------- per-batch 3x3 Kabsch (Jacobi SVD) ----------------
__device__ __forceinline__ float det3f(const float R[3][3]) {
  return R[0][0]*(R[1][1]*R[2][2]-R[1][2]*R[2][1])
       - R[0][1]*(R[1][0]*R[2][2]-R[1][2]*R[2][0])
       + R[0][2]*(R[1][0]*R[2][1]-R[1][1]*R[2][0]);
}

__global__ void finalize_kernel(const float* __restrict__ acc, float* __restrict__ out) {
  int b = threadIdx.x;
  if (b >= B_) return;
  const float* A = acc + b * 16;
  float sa[3] = {A[9], A[10], A[11]};
  float sc[3] = {A[12], A[13], A[14]};
  const float invN = 1.0f / (float)N_;
  float H[3][3];
  for (int i = 0; i < 3; ++i)
    for (int j = 0; j < 3; ++j)
      H[i][j] = A[i * 3 + j] - sa[i] * sc[j] * invN;

  // S = H^T H, Jacobi eigen-decomposition S = V D V^T
  float S[3][3];
  for (int i = 0; i < 3; ++i)
    for (int j = 0; j < 3; ++j)
      S[i][j] = H[0][i]*H[0][j] + H[1][i]*H[1][j] + H[2][i]*H[2][j];
  float V[3][3] = {{1,0,0},{0,1,0},{0,0,1}};
  const int PP[3] = {0, 0, 1}, QQ[3] = {1, 2, 2};
  for (int sweep = 0; sweep < 10; ++sweep) {
    for (int t = 0; t < 3; ++t) {
      int p = PP[t], q = QQ[t];
      float apq = S[p][q];
      if (fabsf(apq) < 1e-20f) continue;
      float tau = (S[q][q] - S[p][p]) / (2.0f * apq);
      float tt  = copysignf(1.0f, tau) / (fabsf(tau) + sqrtf(1.0f + tau * tau));
      float cth = 1.0f / sqrtf(1.0f + tt * tt);
      float sth = tt * cth;
      for (int k = 0; k < 3; ++k) { float u = S[k][p], v = S[k][q]; S[k][p] = cth*u - sth*v; S[k][q] = sth*u + cth*v; }
      for (int k = 0; k < 3; ++k) { float u = S[p][k], v = S[q][k]; S[p][k] = cth*u - sth*v; S[q][k] = sth*u + cth*v; }
      for (int k = 0; k < 3; ++k) { float u = V[k][p], v = V[k][q]; V[k][p] = cth*u - sth*v; V[k][q] = sth*u + cth*v; }
    }
  }
  // sort eigenvalues descending (swap V columns accordingly)
  float dv[3] = {S[0][0], S[1][1], S[2][2]};
  for (int a_ = 0; a_ < 2; ++a_)
    for (int c_ = 0; c_ < 2 - a_; ++c_)
      if (dv[c_] < dv[c_ + 1]) {
        float tmp = dv[c_]; dv[c_] = dv[c_ + 1]; dv[c_ + 1] = tmp;
        for (int k = 0; k < 3; ++k) { float u = V[k][c_]; V[k][c_] = V[k][c_ + 1]; V[k][c_ + 1] = u; }
      }

  // left singular vectors: u_k = normalize(H v_k), Gram-Schmidt, u2 = u0 x u1
  float U[3][3];
  for (int i = 0; i < 3; ++i)
    U[0][i] = H[i][0]*V[0][0] + H[i][1]*V[1][0] + H[i][2]*V[2][0];
  float nrm = sqrtf(U[0][0]*U[0][0] + U[0][1]*U[0][1] + U[0][2]*U[0][2]);
  if (nrm > 1e-12f) { float s = 1.0f / nrm; U[0][0]*=s; U[0][1]*=s; U[0][2]*=s; }
  else { U[0][0] = 1.0f; U[0][1] = 0.0f; U[0][2] = 0.0f; }

  for (int i = 0; i < 3; ++i)
    U[1][i] = H[i][0]*V[0][1] + H[i][1]*V[1][1] + H[i][2]*V[2][1];
  float d01 = U[1][0]*U[0][0] + U[1][1]*U[0][1] + U[1][2]*U[0][2];
  for (int i = 0; i < 3; ++i) U[1][i] -= d01 * U[0][i];
  nrm = sqrtf(U[1][0]*U[1][0] + U[1][1]*U[1][1] + U[1][2]*U[1][2]);
  if (nrm > 1e-12f) { float s = 1.0f / nrm; U[1][0]*=s; U[1][1]*=s; U[1][2]*=s; }
  else {
    float e[3];
    if (fabsf(U[0][0]) < 0.9f) { e[0] = 1.0f; e[1] = 0.0f; e[2] = 0.0f; }
    else                       { e[0] = 0.0f; e[1] = 1.0f; e[2] = 0.0f; }
    U[1][0] = U[0][1]*e[2] - U[0][2]*e[1];
    U[1][1] = U[0][2]*e[0] - U[0][0]*e[2];
    U[1][2] = U[0][0]*e[1] - U[0][1]*e[0];
    float n2 = sqrtf(U[1][0]*U[1][0] + U[1][1]*U[1][1] + U[1][2]*U[1][2]);
    float s = 1.0f / n2; U[1][0]*=s; U[1][1]*=s; U[1][2]*=s;
  }
  U[2][0] = U[0][1]*U[1][2] - U[0][2]*U[1][1];
  U[2][1] = U[0][2]*U[1][0] - U[0][0]*U[1][2];
  U[2][2] = U[0][0]*U[1][1] - U[0][1]*U[1][0];

  float R[3][3];
  for (int i = 0; i < 3; ++i)
    for (int j = 0; j < 3; ++j)
      R[i][j] = V[i][0]*U[0][j] + V[i][1]*U[1][j] + V[i][2]*U[2][j];
  if (det3f(R) < 0.0f)
    for (int i = 0; i < 3; ++i)
      for (int j = 0; j < 3; ++j)
        R[i][j] = V[i][0]*U[0][j] + V[i][1]*U[1][j] - V[i][2]*U[2][j];

  float am[3] = {sa[0]*invN, sa[1]*invN, sa[2]*invN};
  float cm[3] = {sc[0]*invN, sc[1]*invN, sc[2]*invN};
  for (int i = 0; i < 3; ++i)
    for (int j = 0; j < 3; ++j)
      out[b * 9 + i * 3 + j] = R[i][j];
  for (int i = 0; i < 3; ++i)
    out[B_ * 9 + b * 3 + i] = -(R[i][0]*am[0] + R[i][1]*am[1] + R[i][2]*am[2]) + cm[i];
}

// ---------------- launch ----------------
extern "C" void kernel_launch(void* const* d_in, const int* in_sizes, int n_in,
                              void* d_out, int out_size, void* d_ws, size_t ws_size,
                              hipStream_t stream) {
  (void)in_sizes; (void)n_in; (void)out_size; (void)ws_size;
  const float* act_emb = (const float*)d_in[0];
  const float* anc_emb = (const float*)d_in[1];
  const float* act_pts = (const float*)d_in[2];
  const float* anc_pts = (const float*)d_in[3];
  float* out = (float*)d_out;

  char* ws = (char*)d_ws;
  float*    acc = (float*)ws;                                   // 512 f32
  _Float16* Qh  = (_Float16*)(ws + 2048);                       // 8 MB
  _Float16* Kh  = (_Float16*)(ws + 2048 + (size_t)B_*N_*DK_*2); // 8 MB
  float*    Vt  = (float*)(ws + 2048 + (size_t)B_*N_*DK_*2 + (size_t)B_*M_*DK_*2); // 1 MB

  zero_acc_kernel<<<2, 256, 0, stream>>>(acc);
  conv_q_kernel<<<(B_ * N_ * DK_ + 255) / 256, 256, 0, stream>>>(act_emb, Qh);
  conv_k_kernel<<<(B_ * M_ * DK_ + 255) / 256, 256, 0, stream>>>(anc_emb, Kh);
  conv_v_kernel<<<(B_ * M_ + 255) / 256, 256, 0, stream>>>(anc_pts, Vt);
  attn_acc_kernel<<<(B_ * (N_ / 16)) / 8, 256, 0, stream>>>(Qh, Kh, Vt, act_pts, acc);
  finalize_kernel<<<1, 32, 0, stream>>>(acc, out);
}
